// HybridGATModel_86672440033578
// MI455X (gfx1250) — compile-verified, matrix operand order checked by reading, not run on hardware
//
#include <hip/hip_runtime.h>
#include <hip/hip_bf16.h>
#include <math.h>

// ---------------------------------------------------------------------------
// GATv2 (3 layers, H*C = 128/128/32) + mean pool + MLP head for MI455X gfx1250.
// Dense GEMMs use v_wmma_f32_16x16x32_f16 (wave32, 16x32 tile per wave,
// clamped A loads -> no exec divergence in the K loop).
// Edge phase is L2-resident gather/scatter with f32 atomics.
// ---------------------------------------------------------------------------

typedef _Float16 v16h  __attribute__((ext_vector_type(16)));
typedef _Float16 half8 __attribute__((ext_vector_type(8)));
typedef float    v8f   __attribute__((ext_vector_type(8)));

#define NEG_SLOPE 0.2f

__device__ __forceinline__ void atomic_max_f32(float* addr, float val) {
  // bit-trick float max: correct for mixed signs with -inf init
  if (val >= 0.0f) atomicMax((int*)addr, __float_as_int(val));
  else             atomicMin((unsigned int*)addr, __float_as_uint(val));
}

// ---------------- elementwise / utility kernels ----------------------------

__global__ void fill_f32_kernel(float* __restrict__ p, float v, int n) {
  int t = blockIdx.x * blockDim.x + threadIdx.x;
  if (t < n) p[t] = v;
}

__global__ void cvt_f16_kernel(const float* __restrict__ s, _Float16* __restrict__ d, int n) {
  int t = blockIdx.x * blockDim.x + threadIdx.x;
  if (t < n) d[t] = (_Float16)s[t];
}

// W: [K, Ncols] f32 row-major  ->  Wt: [Ncols, K] f16 row-major
__global__ void transpose_f16_kernel(const float* __restrict__ W, _Float16* __restrict__ Wt,
                                     int K, int Ncols) {
  int t = blockIdx.x * blockDim.x + threadIdx.x;
  if (t < K * Ncols) {
    int n = t / K, k = t % K;
    Wt[t] = (_Float16)W[k * Ncols + n];
  }
}

// ---------------- WMMA GEMM: C[M,N] = A[M,K](f16) @ Bt[N,K]^T(f16) + bias ---
// One 16x32 output tile per wave32 (two 16x16 WMMA accumulators sharing the
// A operand); K consumed in chunks of 32.
// A-matrix VGPR layout (ISA 7.12.2, 16-bit A 16x32):
//   lanes 0-15: row=lane,   K = {0..7, 16..23}
//   lanes16-31: row=lane-16,K = {8..15, 24..31}
// B uses the mirrored layout with column-per-lane (Bt rows are B columns).
// C/D: v8f, n = lane&15, m = i + (lane>>4)*8.

__device__ __forceinline__ v16h load_tile16(const _Float16* p) {
  half8 lo = *(const half8*)(p);
  half8 hi = *(const half8*)(p + 16);
  v16h v;
#pragma unroll
  for (int j = 0; j < 8; ++j) { v[j] = lo[j]; v[j + 8] = hi[j]; }
  return v;
}

__global__ void gemm_wmma_kernel(const _Float16* __restrict__ A,   // [M,K]
                                 const _Float16* __restrict__ Bt,  // [N,K]
                                 const float*    __restrict__ bias,// [N]
                                 float*          __restrict__ C,   // [M,N]
                                 int M, int N, int K) {
  int wave = (blockIdx.x * blockDim.x + threadIdx.x) >> 5;
  int lane = threadIdx.x & 31;
  int ntiles = N >> 5;                       // 16x32 tiles along N
  int tileM = wave / ntiles;
  int tileN = wave - tileM * ntiles;
  if (tileM * 16 >= M) return;               // uniform per wave

  int hi    = lane >> 4;                     // 0 or 1 (lane half)
  int r     = lane & 15;
  int arow  = tileM * 16 + r;
  int arow_c = (arow < M) ? arow : (M - 1);  // clamp: OOB rows read valid mem,
                                             // results discarded by store guard
  int brow0 = tileN * 32 + r;                // always < N (N multiple of 32)
  int brow1 = brow0 + 16;

  const _Float16* ap  = A  + (size_t)arow_c * K + hi * 8;
  const _Float16* bp0 = Bt + (size_t)brow0  * K + hi * 8;
  const _Float16* bp1 = Bt + (size_t)brow1  * K + hi * 8;

  v8f acc0 = {}, acc1 = {};
  for (int k0 = 0; k0 < K; k0 += 32) {
    v16h a  = load_tile16(ap  + k0);
    v16h b0 = load_tile16(bp0 + k0);
    v16h b1 = load_tile16(bp1 + k0);
    acc0 = __builtin_amdgcn_wmma_f32_16x16x32_f16(false, a, false, b0,
                                                  (short)0, acc0, false, false);
    acc1 = __builtin_amdgcn_wmma_f32_16x16x32_f16(false, a, false, b1,
                                                  (short)0, acc1, false, false);
  }

  int ncol0 = tileN * 32 + r;
  int ncol1 = ncol0 + 16;
  float bi0 = bias[ncol0], bi1 = bias[ncol1];
#pragma unroll
  for (int i = 0; i < 8; ++i) {
    int m = tileM * 16 + i + hi * 8;
    if (m < M) {
      C[(size_t)m * N + ncol0] = acc0[i] + bi0;
      C[(size_t)m * N + ncol1] = acc1[i] + bi1;
    }
  }
}

// ---------------- edge kernels ---------------------------------------------
// One wave per edge: lanes stripe the C=32 channels per head (coalesced 128B
// loads from L2-resident xl/xr), wave32 shfl_xor reduction for the logit dot.

__global__ void edge_logits_kernel(const float* __restrict__ xl,
                                   const float* __restrict__ xr,
                                   const int*   __restrict__ esrc,
                                   const int*   __restrict__ edst,
                                   const float* __restrict__ att,   // [H,C]
                                   float* __restrict__ elog,        // [Etot,H]
                                   float* __restrict__ lmax,        // [N,H]
                                   int E, int Etot, int H, int C, int HC) {
  int gid = blockIdx.x * blockDim.x + threadIdx.x;
  int e = gid >> 5;
  int lane = threadIdx.x & 31;
  if (e >= Etot) return;
  int s = (e < E) ? esrc[e] : (e - E);     // self-loops appended
  int d = (e < E) ? edst[e] : (e - E);
  for (int h = 0; h < H; ++h) {
    float acc = 0.0f;
    for (int c = lane; c < C; c += 32) {
      float v = xl[(size_t)s * HC + h * C + c] + xr[(size_t)d * HC + h * C + c];
      v = (v > 0.0f) ? v : NEG_SLOPE * v;  // leaky_relu
      acc += v * att[h * C + c];
    }
#pragma unroll
    for (int off = 16; off > 0; off >>= 1) acc += __shfl_xor(acc, off, 32);
    if (lane == 0) {
      elog[(size_t)e * H + h] = acc;
      atomic_max_f32(&lmax[(size_t)d * H + h], acc);
    }
  }
}

__global__ void edge_exp_kernel(const int* __restrict__ edst,
                                float* __restrict__ elog,        // in: logit, out: exp
                                const float* __restrict__ lmax,
                                float* __restrict__ denom,       // [N,H]
                                int E, int Etot, int H) {
  int t = blockIdx.x * blockDim.x + threadIdx.x;
  if (t >= Etot * H) return;
  int e = t / H, h = t - e * H;
  int d = (e < E) ? edst[e] : (e - E);
  float v = __expf(elog[t] - lmax[d * H + h]);
  elog[t] = v;
  atomicAdd(&denom[d * H + h], v);
}

// thread per (edge, h*C+c): alpha-weighted scatter into out[dst]
__global__ void edge_aggr_kernel(const float* __restrict__ xl,
                                 const float* __restrict__ elog,  // exp'd
                                 const float* __restrict__ denom,
                                 const int*   __restrict__ esrc,
                                 const int*   __restrict__ edst,
                                 float* __restrict__ out,         // [N,HC], pre-zeroed
                                 int E, int Etot, int H, int C, int HC) {
  int t = blockIdx.x * blockDim.x + threadIdx.x;
  if (t >= Etot * HC) return;
  int e = t / HC, r = t - e * HC;
  int h = r / C;
  int s = (e < E) ? esrc[e] : (e - E);
  int d = (e < E) ? edst[e] : (e - E);
  float alpha = elog[(size_t)e * H + h] / (denom[(size_t)d * H + h] + 1e-16f);
  atomicAdd(&out[(size_t)d * HC + r], alpha * xl[(size_t)s * HC + r]);
}

__global__ void bias_elu_kernel(float* __restrict__ h, const float* __restrict__ bias,
                                int N, int D) {
  int t = blockIdx.x * blockDim.x + threadIdx.x;
  if (t >= N * D) return;
  float v = h[t] + bias[t % D];
  h[t] = (v > 0.0f) ? v : (__expf(v) - 1.0f);
}

// ---------------- pooling + head -------------------------------------------

__global__ void pool_kernel(const float* __restrict__ h, const int* __restrict__ batch,
                            float* __restrict__ sums, float* __restrict__ cnt,
                            int N, int D) {
  int t = blockIdx.x * blockDim.x + threadIdx.x;
  if (t >= N * D) return;
  int n = t / D, d = t - n * D;
  int b = batch[n];
  atomicAdd(&sums[b * D + d], h[t]);
  if (d == 0) atomicAdd(&cnt[b], 1.0f);
}

__global__ void head_kernel(const float* __restrict__ sums, const float* __restrict__ cnt,
                            const float* __restrict__ meta,      // [B,12]
                            const float* __restrict__ Wh1,       // [44,32]
                            const float* __restrict__ bh1,       // [32]
                            const float* __restrict__ Wh2,       // [32,1]
                            const float* __restrict__ bh2,       // [1]
                            float* __restrict__ out, int B) {
  int b = blockIdx.x * blockDim.x + threadIdx.x;
  if (b >= B) return;
  float c = cnt[b]; if (c < 1.0f) c = 1.0f;
  float z[44];
#pragma unroll
  for (int i = 0; i < 32; ++i) z[i] = sums[b * 32 + i] / c;
#pragma unroll
  for (int i = 0; i < 12; ++i) z[32 + i] = meta[b * 12 + i];
  float o = bh2[0];
  for (int j = 0; j < 32; ++j) {
    float hj = bh1[j];
    for (int k = 0; k < 44; ++k) hj += z[k] * Wh1[k * 32 + j];
    hj = (hj > 0.0f) ? hj : 0.0f;           // relu
    o += hj * Wh2[j];
  }
  out[b] = o;
}

// ---------------------------------------------------------------------------

static inline unsigned nblk(long n, int t) { return (unsigned)((n + t - 1) / t); }

extern "C" void kernel_launch(void* const* d_in, const int* in_sizes, int n_in,
                              void* d_out, int out_size, void* d_ws, size_t ws_size,
                              hipStream_t stream) {
  const float* x     = (const float*)d_in[0];
  const int*   eidx  = (const int*)d_in[1];
  const int*   batch = (const int*)d_in[2];
  const float* meta  = (const float*)d_in[3];

  const int NODE_IN = 128;
  int N = in_sizes[0] / NODE_IN;
  int E = in_sizes[1] / 2;
  int B = in_sizes[3] / 12;
  int Etot = E + N;
  const int* esrc = eidx;
  const int* edst = eidx + E;

  // per-layer parameters (dict order): Wl, bl, Wr, br, att, b
  const float *Wl[3], *bl[3], *Wr[3], *br[3], *att[3], *bb[3];
  for (int l = 0; l < 3; ++l) {
    int base = 4 + 6 * l;
    Wl[l]  = (const float*)d_in[base + 0];
    bl[l]  = (const float*)d_in[base + 1];
    Wr[l]  = (const float*)d_in[base + 2];
    br[l]  = (const float*)d_in[base + 3];
    att[l] = (const float*)d_in[base + 4];
    bb[l]  = (const float*)d_in[base + 5];
  }
  const float* Wh1 = (const float*)d_in[22];
  const float* bh1 = (const float*)d_in[23];
  const float* Wh2 = (const float*)d_in[24];
  const float* bh2 = (const float*)d_in[25];

  // layer configs: {din, H, C}
  const int Ldin[3] = {128, 128, 128};
  const int LH[3]   = {4, 4, 1};
  const int LC[3]   = {32, 32, 32};

  // -------- workspace carve-out (256B aligned) --------
  char* ws = (char*)d_ws;
  size_t off = 0;
  auto carve = [&](size_t bytes) -> char* {
    char* p = ws + off;
    off = (off + bytes + 255) & ~(size_t)255;
    return p;
  };
  float*    hbuf  = (float*)   carve((size_t)N * 128 * 4);   // layer output / next input
  _Float16* a16   = (_Float16*)carve((size_t)N * 128 * 2);   // f16 activations
  _Float16* wl16  = (_Float16*)carve((size_t)128 * 128 * 2); // f16 W^T (lin_l)
  _Float16* wr16  = (_Float16*)carve((size_t)128 * 128 * 2); // f16 W^T (lin_r)
  float*    xlb   = (float*)   carve((size_t)N * 128 * 4);
  float*    xrb   = (float*)   carve((size_t)N * 128 * 4);
  float*    elog  = (float*)   carve((size_t)Etot * 4 * 4);
  float*    lmax  = (float*)   carve((size_t)N * 4 * 4);
  float*    denom = (float*)   carve((size_t)N * 4 * 4);
  float*    sums  = (float*)   carve((size_t)B * 32 * 4);
  float*    cnt   = (float*)   carve((size_t)B * 4);
  (void)ws_size; (void)n_in; (void)out_size;

  const float* hin = x;
  for (int l = 0; l < 3; ++l) {
    int din = Ldin[l], H = LH[l], C = LC[l], HC = H * C;
    int Mtiles = (N + 15) / 16;
    int Ntiles = HC / 32;                    // 16x32 tiles per wave
    long waves = (long)Mtiles * Ntiles;

    cvt_f16_kernel<<<nblk((long)N * din, 256), 256, 0, stream>>>(hin, a16, N * din);
    transpose_f16_kernel<<<nblk((long)din * HC, 256), 256, 0, stream>>>(Wl[l], wl16, din, HC);
    transpose_f16_kernel<<<nblk((long)din * HC, 256), 256, 0, stream>>>(Wr[l], wr16, din, HC);

    gemm_wmma_kernel<<<nblk(waves * 32, 256), 256, 0, stream>>>(a16, wl16, bl[l], xlb, N, HC, din);
    gemm_wmma_kernel<<<nblk(waves * 32, 256), 256, 0, stream>>>(a16, wr16, br[l], xrb, N, HC, din);

    fill_f32_kernel<<<nblk((long)N * H, 256), 256, 0, stream>>>(lmax, -INFINITY, N * H);
    fill_f32_kernel<<<nblk((long)N * H, 256), 256, 0, stream>>>(denom, 0.0f, N * H);
    fill_f32_kernel<<<nblk((long)N * HC, 256), 256, 0, stream>>>(hbuf, 0.0f, N * HC);

    edge_logits_kernel<<<nblk((long)Etot * 32, 256), 256, 0, stream>>>(
        xlb, xrb, esrc, edst, att[l], elog, lmax, E, Etot, H, C, HC);
    edge_exp_kernel<<<nblk((long)Etot * H, 256), 256, 0, stream>>>(
        edst, elog, lmax, denom, E, Etot, H);
    edge_aggr_kernel<<<nblk((long)Etot * HC, 256), 256, 0, stream>>>(
        xlb, elog, denom, esrc, edst, hbuf, E, Etot, H, C, HC);

    // concat layers: out dim = HC; last layer H=1 so head-mean is identity
    bias_elu_kernel<<<nblk((long)N * HC, 256), 256, 0, stream>>>(hbuf, bb[l], N, HC);

    hin = hbuf;
  }

  // global mean pool (D = 32) + MLP head
  fill_f32_kernel<<<nblk((long)B * 32, 256), 256, 0, stream>>>(sums, 0.0f, B * 32);
  fill_f32_kernel<<<nblk((long)B, 256), 256, 0, stream>>>(cnt, 0.0f, B);
  pool_kernel<<<nblk((long)N * 32, 256), 256, 0, stream>>>(hbuf, batch, sums, cnt, N, 32);
  head_kernel<<<nblk((long)B, 256), 256, 0, stream>>>(sums, cnt, meta, Wh1, bh1, Wh2, bh2,
                                                      (float*)d_out, B);
}
